// SatelliteCBF_50251117363655
// MI455X (gfx1250) — matrix operand order repaired and checked by reading, not compiled, not run on hardware
//
#include <hip/hip_runtime.h>
#include <math.h>

typedef __attribute__((ext_vector_type(16))) _Float16 v16h;
typedef __attribute__((ext_vector_type(8)))  _Float16 v8h;
typedef __attribute__((ext_vector_type(8)))  float    v8f;

#define N_DIMS 6
#define HIDDEN 128
#define NHID 4
#define ROWS_PER_ELEM 8      // V row + 6 Jacobian rows + 1 pad row
#define ELEMS_PER_WG 16
#define M_ROWS 128           // ELEMS_PER_WG * ROWS_PER_ELEM
#define THREADS 256          // 8 wave32 waves
#define SA 136               // hidden-layer A stride (halves), padded vs 128 for banks
#define SA0 40               // layer-0 A stride (halves), K=32 + pad

// ---- dynamic LDS layout (in halves, then floats) ----
#define A_HALVES (M_ROWS * SA)              // 17408
#define OFF_A0   0
#define OFF_A1   (A_HALVES)
#define OFF_WIN  (2 * A_HALVES)             // packed w_in B-tiles
#define WIN_HALVES (8 * 32 * 16)            // 1 kt * 8 nt * 32 lanes * 16 halves
#define OFF_WH   (OFF_WIN + WIN_HALVES)
#define WH_LAYER_HALVES (4 * 8 * 32 * 16)   // 4 kt * 8 nt * 32 lanes * 16 halves
#define HALVES_TOTAL (OFF_WH + NHID * WH_LAYER_HALVES)
#define OFF_F_BIN  0
#define OFF_F_BH   (HIDDEN)
#define OFF_F_WOUT (HIDDEN + NHID * HIDDEN)
#define OFF_F_BOUT (OFF_F_WOUT + HIDDEN)
#define FLOATS_TOTAL (OFF_F_BOUT + 1)
#define SMEM_BYTES ((size_t)HALVES_TOTAL * 2 + (size_t)FLOATS_TOTAL * 4)

// Branch-free tanh: hardware V_TANH_F32 on gfx1250 if the builtin exists,
// else a divergence-free native-exp formulation.
__device__ __forceinline__ float fast_tanh(float v) {
#if __has_builtin(__builtin_amdgcn_tanhf)
    return __builtin_amdgcn_tanhf(v);
#elif __has_builtin(__builtin_amdgcn_tanh_f32)
    return __builtin_amdgcn_tanh_f32(v);
#else
    const float e = __expf(2.0f * v);       // v_exp_f32-based, no branches
    return 1.0f - 2.0f / (e + 1.0f);
#endif
}

template<int G>
__device__ __forceinline__ void layer_gemm(_Float16* __restrict__ H,
                                           const float* __restrict__ F,
                                           int lane, int mrow, int koff, int rgrp)
{
    constexpr int  OFF_IN   = (G & 1) ? OFF_A1 : OFF_A0;
    constexpr int  OFF_OUT  = (G & 1) ? OFF_A0 : OFF_A1;
    constexpr int  SAIN     = (G == 0) ? SA0 : SA;
    constexpr int  KSTEPS   = (G == 0) ? 1 : 4;
    constexpr int  OFF_W    = (G == 0) ? OFF_WIN : (OFF_WH + (G - 1) * WH_LAYER_HALVES);
    constexpr int  OFF_B    = (G == 0) ? OFF_F_BIN : (OFF_F_BH + (G - 1) * HIDDEN);
    constexpr bool DO_TANH  = (G < NHID);

    const _Float16* __restrict__ Ain  = H + OFF_IN;
    _Float16*       __restrict__ Aout = H + OFF_OUT;
    const _Float16* __restrict__ W    = H + OFF_W;
    const float*    __restrict__ bias = F + OFF_B;

    v8f acc[8] = {};

    #pragma unroll
    for (int kt = 0; kt < KSTEPS; ++kt) {
        const _Float16* ap = Ain + mrow * SAIN + kt * 32 + koff;
        v8h alo = *(const v8h*)ap;
        v8h ahi = *(const v8h*)(ap + 8);
        v16h a = __builtin_shufflevector(alo, ahi,
                 0,1,2,3,4,5,6,7,8,9,10,11,12,13,14,15);
        #pragma unroll
        for (int nt = 0; nt < 8; ++nt) {
            const _Float16* bp = W + (((kt * 8 + nt) * 32 + lane) << 4);
            v8h blo = *(const v8h*)bp;
            v8h bhi = *(const v8h*)(bp + 8);
            v16h bf = __builtin_shufflevector(blo, bhi,
                      0,1,2,3,4,5,6,7,8,9,10,11,12,13,14,15);
            acc[nt] = __builtin_amdgcn_wmma_f32_16x16x32_f16(
                          false, a, false, bf, (short)0, acc[nt], false, false);
        }
    }

    // bias + tanh + Jacobian scale: lane-local (each lane holds one element's
    // 8-row group for one column in the C fragment).
    #pragma unroll
    for (int nt = 0; nt < 8; ++nt) {
        const int col = nt * 16 + (lane & 15);
        float v0 = acc[nt][0] + bias[col];
        float scale = 1.0f;
        if (DO_TANH) { const float t = fast_tanh(v0); v0 = t; scale = 1.0f - t * t; }
        _Float16* op = Aout + rgrp * SA + col;
        op[0] = (_Float16)v0;
        #pragma unroll
        for (int j = 1; j < 7; ++j) op[j * SA] = (_Float16)(scale * acc[nt][j]);
        op[7 * SA] = (_Float16)0.0f;
    }
}

__global__ __launch_bounds__(THREADS, 1)
void satcbf_vjac_kernel(const float* __restrict__ x,
                        const float* __restrict__ xc,
                        const float* __restrict__ xr,
                        const float* __restrict__ w_in,
                        const float* __restrict__ b_in,
                        const float* __restrict__ w_h,
                        const float* __restrict__ b_h,
                        const float* __restrict__ w_out,
                        const float* __restrict__ b_out,
                        float* __restrict__ out,
                        int bs)
{
    extern __shared__ char smem_raw[];
    _Float16* H = (_Float16*)smem_raw;
    float*    F = (float*)(smem_raw + (size_t)HALVES_TOTAL * 2);

    const int tid   = threadIdx.x;
    const int lane  = tid & 31;
    const int wave  = tid >> 5;
    const int bbase = blockIdx.x * ELEMS_PER_WG;

    // ---------- preload: pack hidden weights into B-fragment order (f16) ----------
    // B[k][n] = w_h[l][n][k]; lane holds 16 contiguous K-halves of one column.
    for (int c = tid; c < NHID * 4 * 8 * 32; c += THREADS) {
        const int l  = c >> 10;
        const int kt = (c >> 8) & 3;
        const int nt = (c >> 5) & 7;
        const int ln = c & 31;
        const int n  = nt * 16 + (ln & 15);
        const int k  = kt * 32 + ((ln >> 4) << 4);
        const float4* s4 = (const float4*)(w_h + ((size_t)(l * HIDDEN + n) * HIDDEN + k));
        const float4 f0 = s4[0], f1 = s4[1], f2 = s4[2], f3 = s4[3];
        _Float16* dst = H + OFF_WH + l * WH_LAYER_HALVES + (((kt * 8 + nt) * 32 + ln) << 4);
        v8h lo = { (_Float16)f0.x, (_Float16)f0.y, (_Float16)f0.z, (_Float16)f0.w,
                   (_Float16)f1.x, (_Float16)f1.y, (_Float16)f1.z, (_Float16)f1.w };
        v8h hi = { (_Float16)f2.x, (_Float16)f2.y, (_Float16)f2.z, (_Float16)f2.w,
                   (_Float16)f3.x, (_Float16)f3.y, (_Float16)f3.z, (_Float16)f3.w };
        *(v8h*)dst = lo;
        *(v8h*)(dst + 8) = hi;
    }
    // w_in (128x6) -> padded 32x128 B-tile set (single K-step)
    {
        const int nt = tid >> 5;
        const int ln = tid & 31;
        const int n  = nt * 16 + (ln & 15);
        const int kb = (ln >> 4) << 4;
        _Float16* dst = H + OFF_WIN + ((nt * 32 + ln) << 4);
        #pragma unroll
        for (int j = 0; j < 16; ++j) {
            const int k = kb + j;
            dst[j] = (k < N_DIMS) ? (_Float16)w_in[n * N_DIMS + k] : (_Float16)0.0f;
        }
    }
    // biases / output weights (f32)
    if (tid < HIDDEN) {
        F[OFF_F_BIN + tid]  = b_in[tid];
        F[OFF_F_WOUT + tid] = w_out[tid];
        #pragma unroll
        for (int l = 0; l < NHID; ++l)
            F[OFF_F_BH + l * HIDDEN + tid] = b_h[l * HIDDEN + tid];
    }
    if (tid == 0) F[OFF_F_BOUT] = b_out[0];

    // ---------- build layer-0 A: rows = [xn ; diag(1/x_range) ; pad] per element ----------
    if (tid < M_ROWS) {
        const int r  = tid;
        const int bl = r >> 3;
        const int cc = r & 7;
        const int b  = bbase + bl;
        _Float16* row = H + OFF_A0 + r * SA0;
        const v8h z = {};
        *(v8h*)(row)      = z;
        *(v8h*)(row + 8)  = z;
        *(v8h*)(row + 16) = z;
        *(v8h*)(row + 24) = z;
        if (b < bs) {
            if (cc == 0) {
                #pragma unroll
                for (int d = 0; d < N_DIMS; ++d)
                    row[d] = (_Float16)((x[b * N_DIMS + d] - xc[d]) / xr[d]);
            } else if (cc <= N_DIMS) {
                row[cc - 1] = (_Float16)(1.0f / xr[cc - 1]);
            }
        }
    }
    __syncthreads();

    // ---------- 5 fused GEMM layers (fully unrolled, constant LDS offsets) ----------
    const int r0   = wave * 16;
    const int mrow = r0 + (lane & 15);      // A-fragment row for this lane
    const int koff = (lane >> 4) * 16;      // A-fragment K sub-offset
    const int rgrp = r0 + (lane >> 4) * 8;  // C-fragment: 8 consecutive M-rows per lane

    layer_gemm<0>(H, F, lane, mrow, koff, rgrp); __syncthreads();
    layer_gemm<1>(H, F, lane, mrow, koff, rgrp); __syncthreads();
    layer_gemm<2>(H, F, lane, mrow, koff, rgrp); __syncthreads();
    layer_gemm<3>(H, F, lane, mrow, koff, rgrp); __syncthreads();
    layer_gemm<4>(H, F, lane, mrow, koff, rgrp); __syncthreads();

    // ---------- output layer: 1x128 dot per row ----------
    if (tid < M_ROWS) {
        const int r  = tid;
        const int bl = r >> 3;
        const int cc = r & 7;
        const int b  = bbase + bl;
        if (b < bs && cc < 7) {
            const _Float16* row = H + OFF_A1 + r * SA;  // GEMM 4 wrote buffer 1
            float accv = 0.0f;
            #pragma unroll
            for (int h0 = 0; h0 < HIDDEN; h0 += 8) {
                const v8h rv = *(const v8h*)(row + h0);
                #pragma unroll
                for (int j = 0; j < 8; ++j)
                    accv += (float)rv[j] * F[OFF_F_WOUT + h0 + j];
            }
            if (cc == 0) out[b] = accv + F[OFF_F_BOUT];
            else         out[bs + b * N_DIMS + (cc - 1)] = accv;
        }
    }
}

extern "C" void kernel_launch(void* const* d_in, const int* in_sizes, int n_in,
                              void* d_out, int out_size, void* d_ws, size_t ws_size,
                              hipStream_t stream) {
    const float* x     = (const float*)d_in[0];
    const float* xc    = (const float*)d_in[1];
    const float* xr    = (const float*)d_in[2];
    const float* w_in  = (const float*)d_in[3];
    const float* b_in  = (const float*)d_in[4];
    const float* w_h   = (const float*)d_in[5];
    const float* b_h   = (const float*)d_in[6];
    const float* w_out = (const float*)d_in[7];
    const float* b_out = (const float*)d_in[8];
    float* out = (float*)d_out;

    const int bs = in_sizes[0] / N_DIMS;
    const int grid = (bs + ELEMS_PER_WG - 1) / ELEMS_PER_WG;

    satcbf_vjac_kernel<<<grid, THREADS, SMEM_BYTES, stream>>>(
        x, xc, xr, w_in, b_in, w_h, b_h, w_out, b_out, out, bs);
}
